// Ligand_Graph_Model_70583492542731
// MI455X (gfx1250) — matrix operand order repaired and checked by reading, not compiled, not run on hardware
//
#include <hip/hip_runtime.h>
#include <stdint.h>

#define NNODE   20000
#define NEDGE   320000
#define NEF     (NEDGE + NNODE)
#define NGRAPH  64
#define IN_DIM  23
#define KPAD0   32
#define EDGE_DIM 6
#define HDIM    256
#define NLAYER  4
#define NEGS    0.2f
#define LNEPS   1e-5f

typedef __attribute__((ext_vector_type(16))) __bf16        v16bf;
typedef __attribute__((ext_vector_type(8)))  float         v8f;
typedef __attribute__((ext_vector_type(4)))  unsigned int  u32x4;
typedef unsigned short u16;
typedef unsigned int   u32;

// ---------- helpers ----------
static __device__ __forceinline__ u16 f2bf(float f) {
  u32 u = __float_as_uint(f);
  u32 r = u + 0x7FFFu + ((u >> 16) & 1u);   // round-to-nearest-even
  return (u16)(r >> 16);
}
// order-preserving float<->uint for atomic max (init 0 == below any real float)
static __device__ __forceinline__ u32 ford(float f) {
  u32 u = __float_as_uint(f);
  return (u & 0x80000000u) ? ~u : (u | 0x80000000u);
}
static __device__ __forceinline__ float forddec(u32 u) {
  return __uint_as_float((u & 0x80000000u) ? (u ^ 0x80000000u) : ~u);
}

// ---------- input prep ----------
__global__ void k_fill_xb(const float* __restrict__ x, u16* __restrict__ xb) {
  int i = blockIdx.x * blockDim.x + threadIdx.x;
  if (i >= NNODE * IN_DIM) return;
  int v = i / IN_DIM, c = i % IN_DIM;
  xb[(size_t)v * KPAD0 + c] = f2bf(x[i]);
}

// W [K x Hout] row-major fp32 -> WT [Hout x Kpad] bf16 (zero padded K)
__global__ void k_transpose_w(const float* __restrict__ W, u16* __restrict__ WT,
                              int K, int Hout, int Kpad) {
  int i = blockIdx.x * blockDim.x + threadIdx.x;
  if (i >= Hout * Kpad) return;
  int n = i / Kpad, k = i % Kpad;
  WT[i] = (k < K) ? f2bf(W[(size_t)k * Hout + n]) : (u16)0;
}

// degree + summed incoming edge_attr (for self-loop 'mean' fill)
__global__ void k_deg(const int* __restrict__ dst, const float* __restrict__ eattr,
                      float* __restrict__ deg, float* __restrict__ lattr) {
  int e = blockIdx.x * blockDim.x + threadIdx.x;
  if (e >= NEDGE) return;
  int d = dst[e];
  atomicAdd(&deg[d], 1.0f);
  #pragma unroll
  for (int c = 0; c < EDGE_DIM; ++c)
    atomicAdd(&lattr[(size_t)d * EDGE_DIM + c], eattr[(size_t)e * EDGE_DIM + c]);
}

__global__ void k_loopdiv(float* __restrict__ lattr, const float* __restrict__ deg) {
  int i = blockIdx.x * blockDim.x + threadIdx.x;
  if (i >= NNODE * EDGE_DIM) return;
  lattr[i] /= fmaxf(deg[i / EDGE_DIM], 1.0f);
}

// ---------- bf16 WMMA GEMM: C[M x 256] = A[M x K]_bf16 @ B + bias ----------
// A row-major bf16; BT is B transposed, [256 x K] bf16. Nout must be 256.
// grid = (M/16), block = 128 (4 waves). Each wave computes a 16x64 strip
// (4 N-tiles): the A fragment is loaded once per k-step and feeds 4 WMMAs.
// K is compile-time so the whole k-loop fully unrolls and the scheduler can
// software-pipeline loads across WMMAs with partial loadcnt waits.
template <int K>
__global__ void gemm_bf16_wmma(const u16* __restrict__ A, const u16* __restrict__ BT,
                               const float* __restrict__ bias, float* __restrict__ C,
                               int Nout) {
  const int lane   = threadIdx.x & 31;
  const int wave   = threadIdx.x >> 5;
  const int laneHi = lane >> 4;      // 0: lanes 0-15, 1: lanes 16-31
  const int lrow   = lane & 15;
  const int tileM  = blockIdx.x;

  const u16* arow  = A  + (size_t)(tileM * 16 + lrow) * K;      // A row = M index
  const u16* bbase = BT + (size_t)(wave * 64 + lrow) * K;       // first of 4 N-tiles

  union FragU { u32x4 u[2]; v16bf v; };
  v8f cacc[4] = {};
  #pragma unroll
  for (int k0 = 0; k0 < K; k0 += 32) {
    // A 16-bit 16x32 layout: laneLo K = {0..7,16..23}, laneHi K = {8..15,24..31}
    FragU af;
    af.u[0] = *(const u32x4*)(arow + k0 + laneHi * 8);
    af.u[1] = *(const u32x4*)(arow + k0 + 16 + laneHi * 8);
    // B 16-bit 32x16 layout: laneLo K = {0..7,16..23}, laneHi K = {8..15,24..31}
    FragU bf[4];
    #pragma unroll
    for (int t = 0; t < 4; ++t) {
      const u32x4* bp = (const u32x4*)(bbase + (size_t)t * 16 * K + k0 + laneHi * 8);
      bf[t].u[0] = bp[0];
      bf[t].u[1] = bp[2];
    }
    #pragma unroll
    for (int t = 0; t < 4; ++t)
      cacc[t] = __builtin_amdgcn_wmma_f32_16x16x32_bf16(false, af.v, false, bf[t].v,
                                                        (short)0, cacc[t], false, false);
  }
  // C/D: lane col = lane&15, VGPR r -> row r + 8*laneHi
  float* cbase = C + (size_t)(tileM * 16 + laneHi * 8) * Nout;
  #pragma unroll
  for (int t = 0; t < 4; ++t) {
    const int col = (wave * 4 + t) * 16 + lrow;
    const float bv = bias ? bias[col] : 0.0f;
    #pragma unroll
    for (int r = 0; r < 8; ++r)
      cbase[(size_t)r * Nout + col] = cacc[t][r] + bv;
  }
}

// ---------- edge pass 1: fused e_emb + leaky_relu + dot(att) + segment max ----------
__global__ void k_edge_logit(const float* __restrict__ hl, const float* __restrict__ hr,
                             const int* __restrict__ src, const int* __restrict__ dst,
                             const float* __restrict__ eattr, const float* __restrict__ lattr,
                             const float* __restrict__ We, const float* __restrict__ att,
                             float* __restrict__ logits, u32* __restrict__ maxb) {
  __shared__ float sWe[EDGE_DIM * HDIM];
  __shared__ float sAtt[HDIM];
  int tid = threadIdx.x;
  for (int j = tid; j < EDGE_DIM * HDIM; j += 256) sWe[j] = We[j];
  if (tid < HDIM) sAtt[tid] = att[tid];
  __syncthreads();

  int e = blockIdx.x * 8 + (tid >> 5);
  if (e >= NEF) return;
  int lane = tid & 31;
  int s, d; const float* a;
  if (e < NEDGE) { s = src[e]; d = dst[e]; a = eattr + (size_t)e * EDGE_DIM; }
  else           { s = d = e - NEDGE;      a = lattr + (size_t)s * EDGE_DIM; }
  float a0 = a[0], a1 = a[1], a2 = a[2], a3 = a[3], a4 = a[4], a5 = a[5];
  const float* pl = hl + (size_t)s * HDIM;
  const float* pr = hr + (size_t)d * HDIM;

  float acc = 0.f;
  #pragma unroll
  for (int t = 0; t < HDIM / 32; ++t) {
    int j = lane + t * 32;
    float ee = a0 * sWe[j]            + a1 * sWe[HDIM + j]     + a2 * sWe[2 * HDIM + j]
             + a3 * sWe[3 * HDIM + j] + a4 * sWe[4 * HDIM + j] + a5 * sWe[5 * HDIM + j];
    float m = pl[j] + pr[j] + ee;
    m = m > 0.f ? m : NEGS * m;        // leaky_relu
    acc += m * sAtt[j];
  }
  #pragma unroll
  for (int o = 16; o > 0; o >>= 1) acc += __shfl_xor(acc, o, 32);
  if (lane == 0) {
    logits[e] = acc;
    atomicMax(&maxb[d], ford(acc));
  }
}

// ---------- edge pass 2: exp + segment sum ----------
__global__ void k_edge_exp(const float* __restrict__ logits, const u32* __restrict__ maxb,
                           const int* __restrict__ dst,
                           float* __restrict__ expv, float* __restrict__ sumb) {
  int e = blockIdx.x * blockDim.x + threadIdx.x;
  if (e >= NEF) return;
  int d = (e < NEDGE) ? dst[e] : e - NEDGE;
  float w = __expf(logits[e] - forddec(maxb[d]));
  expv[e] = w;
  atomicAdd(&sumb[d], w);
}

// ---------- edge pass 3: unnormalized weighted scatter ----------
__global__ void k_edge_scatter(const float* __restrict__ hl, const float* __restrict__ expv,
                               const int* __restrict__ src, const int* __restrict__ dst,
                               float* __restrict__ accb) {
  int e = blockIdx.x * 8 + (threadIdx.x >> 5);
  if (e >= NEF) return;
  int lane = threadIdx.x & 31;
  int s, d;
  if (e < NEDGE) { s = src[e]; d = dst[e]; } else { s = d = e - NEDGE; }
  float w = expv[e];
  const float* pl = hl + (size_t)s * HDIM;
  float* pa = accb + (size_t)d * HDIM;
  #pragma unroll
  for (int t = 0; t < HDIM / 32; ++t) {
    int j = lane + t * 32;
    atomicAdd(&pa[j], pl[j] * w);
  }
}

// ---------- node finalize: normalize softmax, +bias, layernorm, relu, ->bf16 ----------
__global__ void k_node_fin(const float* __restrict__ accb, const float* __restrict__ sumb,
                           const float* __restrict__ bias, const float* __restrict__ lnw,
                           const float* __restrict__ lnb, u16* __restrict__ hb) {
  __shared__ float red[HDIM];
  int v = blockIdx.x, j = threadIdx.x;
  float val = accb[(size_t)v * HDIM + j] / sumb[v] + bias[j];
  red[j] = val; __syncthreads();
  for (int s = HDIM / 2; s > 0; s >>= 1) { if (j < s) red[j] += red[j + s]; __syncthreads(); }
  float mu = red[0] * (1.0f / HDIM);
  __syncthreads();
  float dv = val - mu;
  red[j] = dv * dv; __syncthreads();
  for (int s = HDIM / 2; s > 0; s >>= 1) { if (j < s) red[j] += red[j + s]; __syncthreads(); }
  float var = red[0] * (1.0f / HDIM);
  float y = dv * rsqrtf(var + LNEPS) * lnw[j] + lnb[j];
  y = fmaxf(y, 0.f);
  hb[(size_t)v * HDIM + j] = f2bf(y);
}

// ---------- pooling ----------
__global__ void k_pool(const float* __restrict__ hout, const int* __restrict__ batch,
                       float* __restrict__ pooled, float* __restrict__ cnt) {
  int i = blockIdx.x * blockDim.x + threadIdx.x;
  if (i >= NNODE * HDIM) return;
  int v = i / HDIM, j = i % HDIM;
  int g = batch[v];
  atomicAdd(&pooled[(size_t)g * HDIM + j], hout[i]);
  if (j == 0) atomicAdd(&cnt[g], 1.0f);
}

__global__ void k_pooldiv(const float* __restrict__ pooled, const float* __restrict__ cnt,
                          float* __restrict__ out) {
  int i = blockIdx.x * blockDim.x + threadIdx.x;
  if (i >= NGRAPH * HDIM) return;
  out[i] = pooled[i] / fmaxf(cnt[i / HDIM], 1.0f);
}

// ---------- host ----------
extern "C" void kernel_launch(void* const* d_in, const int* in_sizes, int n_in,
                              void* d_out, int out_size, void* d_ws, size_t ws_size,
                              hipStream_t stream) {
  (void)in_sizes; (void)n_in; (void)out_size; (void)ws_size;
  const float* x     = (const float*)d_in[0];
  const int*   ei    = (const int*)  d_in[1];
  const float* eattr = (const float*)d_in[2];
  const int*   batch = (const int*)  d_in[3];
  const float* W_l0  = (const float*)d_in[4];
  const float* W_r0  = (const float*)d_in[5];
  const float* W_l   = (const float*)d_in[6];
  const float* W_r   = (const float*)d_in[7];
  const float* b_l   = (const float*)d_in[8];
  const float* b_r   = (const float*)d_in[9];
  const float* W_e   = (const float*)d_in[10];
  const float* attW  = (const float*)d_in[11];
  const float* gbias = (const float*)d_in[12];
  const float* ln_w  = (const float*)d_in[13];
  const float* ln_b  = (const float*)d_in[14];
  const float* lin_W = (const float*)d_in[15];
  const float* lin_b = (const float*)d_in[16];
  float* out = (float*)d_out;
  const int* srcArr = ei;
  const int* dstArr = ei + NEDGE;

  char* base = (char*)d_ws;
  size_t off = 0;
  auto alloc = [&](size_t bytes) -> void* {
    void* p = base + off;
    off = (off + bytes + 255) & ~(size_t)255;
    return p;
  };
  u16*   xb    = (u16*)  alloc((size_t)NNODE * KPAD0 * 2);
  u16*   w0lT  = (u16*)  alloc((size_t)HDIM * KPAD0 * 2);
  u16*   w0rT  = (u16*)  alloc((size_t)HDIM * KPAD0 * 2);
  u16*   wlT   = (u16*)  alloc((size_t)(NLAYER - 1) * HDIM * HDIM * 2);
  u16*   wrT   = (u16*)  alloc((size_t)(NLAYER - 1) * HDIM * HDIM * 2);
  u16*   linT  = (u16*)  alloc((size_t)HDIM * HDIM * 2);
  u16*   hb    = (u16*)  alloc((size_t)NNODE * HDIM * 2);
  float* hl    = (float*)alloc((size_t)NNODE * HDIM * 4);
  float* hr    = (float*)alloc((size_t)NNODE * HDIM * 4);
  float* accb  = (float*)alloc((size_t)NNODE * HDIM * 4);
  float* logit = (float*)alloc((size_t)NEF * 4);
  float* expv  = (float*)alloc((size_t)NEF * 4);
  u32*   maxb  = (u32*)  alloc((size_t)NNODE * 4);
  float* sumb  = (float*)alloc((size_t)NNODE * 4);
  float* deg   = (float*)alloc((size_t)NNODE * 4);
  float* lattr = (float*)alloc((size_t)NNODE * EDGE_DIM * 4);
  float* pooled= (float*)alloc((size_t)NGRAPH * HDIM * 4);
  float* cnt   = (float*)alloc((size_t)NGRAPH * 4);

  // --- input conversion / weight prep ---
  hipMemsetAsync(xb, 0, (size_t)NNODE * KPAD0 * 2, stream);
  k_fill_xb<<<(NNODE * IN_DIM + 255) / 256, 256, 0, stream>>>(x, xb);
  k_transpose_w<<<(HDIM * KPAD0 + 255) / 256, 256, 0, stream>>>(W_l0, w0lT, IN_DIM, HDIM, KPAD0);
  k_transpose_w<<<(HDIM * KPAD0 + 255) / 256, 256, 0, stream>>>(W_r0, w0rT, IN_DIM, HDIM, KPAD0);
  for (int i = 0; i < NLAYER - 1; ++i) {
    k_transpose_w<<<(HDIM * HDIM + 255) / 256, 256, 0, stream>>>(
        W_l + (size_t)i * HDIM * HDIM, wlT + (size_t)i * HDIM * HDIM, HDIM, HDIM, HDIM);
    k_transpose_w<<<(HDIM * HDIM + 255) / 256, 256, 0, stream>>>(
        W_r + (size_t)i * HDIM * HDIM, wrT + (size_t)i * HDIM * HDIM, HDIM, HDIM, HDIM);
  }
  k_transpose_w<<<(HDIM * HDIM + 255) / 256, 256, 0, stream>>>(lin_W, linT, HDIM, HDIM, HDIM);

  // --- self-loop attr = mean of incoming edge_attr ---
  hipMemsetAsync(deg, 0, (size_t)NNODE * 4, stream);
  hipMemsetAsync(lattr, 0, (size_t)NNODE * EDGE_DIM * 4, stream);
  k_deg<<<(NEDGE + 255) / 256, 256, 0, stream>>>(dstArr, eattr, deg, lattr);
  k_loopdiv<<<(NNODE * EDGE_DIM + 255) / 256, 256, 0, stream>>>(lattr, deg);

  dim3 ggrid(NNODE / 16);  // 1250 blocks, 4 waves each; wave covers 16x64 of C
  for (int i = 0; i < NLAYER; ++i) {
    if (i == 0) {
      gemm_bf16_wmma<KPAD0><<<ggrid, 128, 0, stream>>>(xb, w0lT, b_l, hl, HDIM);
      gemm_bf16_wmma<KPAD0><<<ggrid, 128, 0, stream>>>(xb, w0rT, b_r, hr, HDIM);
    } else {
      gemm_bf16_wmma<HDIM><<<ggrid, 128, 0, stream>>>(hb, wlT + (size_t)(i - 1) * HDIM * HDIM,
                                                      b_l + (size_t)i * HDIM, hl, HDIM);
      gemm_bf16_wmma<HDIM><<<ggrid, 128, 0, stream>>>(hb, wrT + (size_t)(i - 1) * HDIM * HDIM,
                                                      b_r + (size_t)i * HDIM, hr, HDIM);
    }
    hipMemsetAsync(maxb, 0, (size_t)NNODE * 4, stream);
    hipMemsetAsync(sumb, 0, (size_t)NNODE * 4, stream);
    hipMemsetAsync(accb, 0, (size_t)NNODE * HDIM * 4, stream);
    k_edge_logit<<<(NEF + 7) / 8, 256, 0, stream>>>(
        hl, hr, srcArr, dstArr, eattr, lattr,
        W_e + (size_t)i * EDGE_DIM * HDIM, attW + (size_t)i * HDIM, logit, maxb);
    k_edge_exp<<<(NEF + 255) / 256, 256, 0, stream>>>(logit, maxb, dstArr, expv, sumb);
    k_edge_scatter<<<(NEF + 7) / 8, 256, 0, stream>>>(hl, expv, srcArr, dstArr, accb);
    k_node_fin<<<NNODE, 256, 0, stream>>>(accb, sumb, gbias + (size_t)i * HDIM,
                                          ln_w + (size_t)i * HDIM, ln_b + (size_t)i * HDIM, hb);
  }

  // final projection (reuse hl as output) + mean pool
  gemm_bf16_wmma<HDIM><<<ggrid, 128, 0, stream>>>(hb, linT, lin_b, hl, HDIM);
  hipMemsetAsync(pooled, 0, (size_t)NGRAPH * HDIM * 4, stream);
  hipMemsetAsync(cnt, 0, (size_t)NGRAPH * 4, stream);
  k_pool<<<(NNODE * HDIM + 255) / 256, 256, 0, stream>>>(hl, batch, pooled, cnt);
  k_pooldiv<<<(NGRAPH * HDIM + 255) / 256, 256, 0, stream>>>(pooled, cnt, out);
}
